// GINO_51616916963789
// MI455X (gfx1250) — matrix operand order repaired and checked by reading, not compiled, not run on hardware
//
#include <hip/hip_runtime.h>
#include <hip/hip_bf16.h>
#include <math.h>

typedef __attribute__((ext_vector_type(16))) _Float16 v16h;
typedef __attribute__((ext_vector_type(8)))  _Float16 v8h;
typedef __attribute__((ext_vector_type(8)))  float    v8f;

#define TWO_PI 6.283185307179586f

__device__ __forceinline__ float gelu_f(float x) {
    // jax.nn.gelu default (approximate=True, tanh form)
    float x3 = x * x * x;
    return 0.5f * x * (1.0f + tanhf(0.7978845608028654f * (x + 0.044715f * x3)));
}

__device__ __forceinline__ v16h cat8(v8h lo, v8h hi) {
    return __builtin_shufflevector(lo, hi, 0, 1, 2, 3, 4, 5, 6, 7,
                                           8, 9, 10, 11, 12, 13, 14, 15);
}

// ---------------------------------------------------------------------------
// WMMA GEMM:  C = act(A @ W + bias)
//   A  : [M, Kd] f16 row-major (Kd mult of 32, M mult of 16)
//   Wp : weights pre-packed in WMMA B-fragment order:
//        Wp[((kchunk*ntiles + ntile)*32 + lane)*16 + j]
//          = W[kchunk*32 + (lane>>4)*16 + j][ntile*16 + (lane&15)]
//        -> each lane loads two contiguous v8h per (kchunk, ntile): fully
//           coalesced 1KB/wave, no strided u16 loads.
//   One wave computes a 32x32 tile (2x2 subtiles, 4 WMMAs per K-step) for
//   ~16 f16-FLOP per fragment byte. Second row-subtile guarded by a
//   wave-uniform predicate (EXEC all-ones around WMMA per ISA 7.12).
// ---------------------------------------------------------------------------
__global__ void k_gemm(const _Float16* __restrict__ A,
                       const _Float16* __restrict__ Wp,
                       const float* __restrict__ bias,
                       _Float16* __restrict__ Ch,
                       float* __restrict__ Cf,
                       int M, int N, int Kd, int n_true, int dogelu) {
    int wave = threadIdx.x >> 5;
    int lane = threadIdx.x & 31;
    int m0 = (blockIdx.x * 4 + wave) * 32;
    int n0 = blockIdx.y * 32;
    if (m0 >= M) return;                       // wave-uniform
    bool m1ok = (m0 + 16) < M;                 // wave-uniform

    int lrow = lane & 15;
    int hsel = lane >> 4;                      // 0: K{0..15}, 1: K{16..31}
    const _Float16* arow0 = A + (size_t)(m0 + lrow) * Kd + hsel * 16;
    const _Float16* arow1 = m1ok ? (arow0 + (size_t)16 * Kd) : arow0;

    // packed-B bases for the two n-subtiles (kchunk stride = N*32 halves)
    const _Float16* wp0 = Wp + ((((size_t)(n0 >> 4)) * 32 + lane) << 4);
    const _Float16* wp1 = wp0 + 512;           // next ntile: 32 lanes * 16 halves
    size_t kstride = (size_t)N * 32;

    v8f c00 = {0.f,0.f,0.f,0.f,0.f,0.f,0.f,0.f};
    v8f c01 = c00, c10 = c00, c11 = c00;

    for (int kc = 0; kc < Kd; kc += 32) {
        v16h a0 = cat8(*(const v8h*)(arow0 + kc), *(const v8h*)(arow0 + kc + 8));
        v16h a1 = cat8(*(const v8h*)(arow1 + kc), *(const v8h*)(arow1 + kc + 8));
        size_t ko = (size_t)(kc >> 5) * kstride;
        v16h b0 = cat8(*(const v8h*)(wp0 + ko), *(const v8h*)(wp0 + ko + 8));
        v16h b1 = cat8(*(const v8h*)(wp1 + ko), *(const v8h*)(wp1 + ko + 8));
        c00 = __builtin_amdgcn_wmma_f32_16x16x32_f16(false, a0, false, b0, (short)0, c00, false, false);
        c01 = __builtin_amdgcn_wmma_f32_16x16x32_f16(false, a0, false, b1, (short)0, c01, false, false);
        c10 = __builtin_amdgcn_wmma_f32_16x16x32_f16(false, a1, false, b0, (short)0, c10, false, false);
        c11 = __builtin_amdgcn_wmma_f32_16x16x32_f16(false, a1, false, b1, (short)0, c11, false, false);
    }

    int col0 = n0 + lrow, col1 = n0 + 16 + lrow;
    float b0c = bias ? bias[col0] : 0.0f;
    float b1c = bias ? bias[col1] : 0.0f;
#pragma unroll
    for (int r = 0; r < 8; ++r) {
        int row0 = m0 + r + hsel * 8;          // C/D layout: VGPR r -> M=r / M=8+r
        float v00 = c00[r] + b0c, v01 = c01[r] + b1c;
        if (dogelu) { v00 = gelu_f(v00); v01 = gelu_f(v01); }
        if (Ch) {
            Ch[(size_t)row0 * N + col0] = (_Float16)v00;
            Ch[(size_t)row0 * N + col1] = (_Float16)v01;
        }
        if (Cf) {
            if (col0 < n_true) Cf[(size_t)row0 * n_true + col0] = v00;
            if (col1 < n_true) Cf[(size_t)row0 * n_true + col1] = v01;
        }
        if (m1ok) {
            int row1 = row0 + 16;
            float v10 = c10[r] + b0c, v11 = c11[r] + b1c;
            if (dogelu) { v10 = gelu_f(v10); v11 = gelu_f(v11); }
            if (Ch) {
                Ch[(size_t)row1 * N + col0] = (_Float16)v10;
                Ch[(size_t)row1 * N + col1] = (_Float16)v11;
            }
            if (Cf) {
                if (col0 < n_true) Cf[(size_t)row1 * n_true + col0] = v10;
                if (col1 < n_true) Cf[(size_t)row1 * n_true + col1] = v11;
            }
        }
    }
}

// ------------- weight pack: f32 [K,N] -> padded f16 WMMA-B fragments -------
__global__ void k_packw(const float* __restrict__ src, _Float16* __restrict__ dst,
                        int K, int N, int Kp, int Np) {
    int t = blockIdx.x * blockDim.x + threadIdx.x;   // Kp*Np
    if (t >= Kp * Np) return;
    int j = t & 15;
    int lane = (t >> 4) & 31;
    int tile = t >> 9;
    int ntiles = Np >> 4;
    int ntile = tile % ntiles;
    int kchunk = tile / ntiles;
    int col = (ntile << 4) + (lane & 15);
    int krow = (kchunk << 5) + ((lane >> 4) << 4) + j;
    dst[t] = (krow < K && col < N) ? (_Float16)src[(size_t)krow * N + col]
                                   : (_Float16)0.0f;
}
__global__ void k_padb(const float* __restrict__ src, float* __restrict__ dst, int N, int Np) {
    int t = blockIdx.x * blockDim.x + threadIdx.x;
    if (t >= Np) return;
    dst[t] = (t < N) ? src[t] : 0.0f;
}

// --------------------------- embeddings & lift -----------------------------
__global__ void k_embed(const float* __restrict__ coords, _Float16* __restrict__ emb, int N) {
    int t = blockIdx.x * blockDim.x + threadIdx.x;   // N*48 threads: (n, d, i<16)
    if (t >= N * 48) return;
    int i = t % 16, d = (t / 16) % 3, n = t / 48;
    float cd = coords[n * 3 + d];
    float freq = powf(10000.0f, (float)i / 16.0f);
    float s, co; __sincosf(cd * freq, &s, &co);
    emb[(size_t)n * 96 + d * 32 + i]      = (_Float16)s;
    emb[(size_t)n * 96 + d * 32 + 16 + i] = (_Float16)co;
}

__global__ void k_lift(const float* __restrict__ x, const float* __restrict__ w,
                       const float* __restrict__ b, float* __restrict__ f, int N) {
    int t = blockIdx.x * blockDim.x + threadIdx.x;   // N*64
    if (t >= N * 64) return;
    int c = t & 63; int n = t >> 6;
    f[t] = x[n * 3 + 0] * w[c] + x[n * 3 + 1] * w[64 + c] + x[n * 3 + 2] * w[128 + c] + b[c];
}

// --------------------------- GNO gather / reduce ---------------------------
__global__ void k_gather_in(const _Float16* __restrict__ ye, const _Float16* __restrict__ xe,
                            const int* __restrict__ idx, _Float16* __restrict__ EF,
                            int p0, int pn) {
    int t = blockIdx.x * blockDim.x + threadIdx.x;   // pn*10*192
    if (t >= pn * 10 * 192) return;
    int col = t % 192, row = t / 192;
    int pl = row / 10, k = row % 10, p = p0 + pl;
    EF[(size_t)row * 192 + col] =
        (col < 96) ? ye[(size_t)idx[(size_t)p * 10 + k] * 96 + col]
                   : xe[(size_t)p * 96 + (col - 96)];
}

__global__ void k_gather_out(const _Float16* __restrict__ xe, const _Float16* __restrict__ oe,
                             const int* __restrict__ idx, _Float16* __restrict__ EF,
                             int q0, int qn) {
    int t = blockIdx.x * blockDim.x + threadIdx.x;
    if (t >= qn * 10 * 192) return;
    int col = t % 192, row = t / 192;
    int ql = row / 10, k = row % 10, q = q0 + ql;
    EF[(size_t)row * 192 + col] =
        (col < 96) ? xe[(size_t)idx[(size_t)q * 10 + k] * 96 + col]
                   : oe[(size_t)q * 96 + (col - 96)];
}

__global__ void k_reduce_in(const float* __restrict__ K3, const float* __restrict__ f,
                            const int* __restrict__ idx, float* __restrict__ h32,
                            _Float16* __restrict__ h16, int p0, int pn) {
    int t = blockIdx.x * blockDim.x + threadIdx.x;   // pn*64
    if (t >= pn * 64) return;
    int c = t & 63; int pl = t >> 6; int p = p0 + pl;
    float acc = 0.f;
#pragma unroll
    for (int k = 0; k < 10; ++k)
        acc += K3[((size_t)pl * 10 + k) * 64 + c] * f[(size_t)idx[(size_t)p * 10 + k] * 64 + c];
    acc *= 0.1f;                                     // mean reduction
    h32[(size_t)p * 64 + c] = acc;
    h16[(size_t)p * 64 + c] = (_Float16)acc;
}

__global__ void k_reduce_out(const float* __restrict__ K3, const float* __restrict__ h,
                             const int* __restrict__ idx, _Float16* __restrict__ oev16,
                             int q0, int qn) {
    int t = blockIdx.x * blockDim.x + threadIdx.x;
    if (t >= qn * 64) return;
    int c = t & 63; int ql = t >> 6; int q = q0 + ql;
    float acc = 0.f;
#pragma unroll
    for (int k = 0; k < 10; ++k)
        acc += K3[((size_t)ql * 10 + k) * 64 + c] * h[(size_t)idx[(size_t)q * 10 + k] * 64 + c];
    oev16[(size_t)q * 64 + c] = (_Float16)acc;       // sum reduction
}

// ------------------ truncated separable DFT (spectral conv) ----------------
// Only the kept corner modes are computed: kx,ky in {0..7} U {24..31}, kz in {0..7}.
__global__ void k_dft_z(const float* __restrict__ h, float2* __restrict__ Fz) {
    int t = blockIdx.x * blockDim.x + threadIdx.x;   // 32*32*8*64
    if (t >= 32 * 32 * 8 * 64) return;
    int c = t & 63, kz = (t >> 6) & 7, y = (t >> 9) & 31, x = t >> 14;
    float re = 0.f, im = 0.f;
    for (int z = 0; z < 32; ++z) {
        float v = h[((((x * 32 + y) * 32 + z)) << 6) + c];
        float s, co; __sincosf(-(TWO_PI / 32.f) * (float)(kz * z), &s, &co);
        re += v * co; im += v * s;
    }
    Fz[t] = make_float2(re, im);
}

__global__ void k_dft_y(const float2* __restrict__ Fz, float2* __restrict__ Fy) {
    int t = blockIdx.x * blockDim.x + threadIdx.x;   // 32*16*8*64
    if (t >= 32 * 16 * 8 * 64) return;
    int c = t & 63, kz = (t >> 6) & 7, ky = (t >> 9) & 15, x = t >> 13;
    int kyA = (ky < 8) ? ky : ky + 16;
    float re = 0.f, im = 0.f;
    for (int y = 0; y < 32; ++y) {
        float2 v = Fz[((size_t)(x * 32 + y) * 8 + kz) * 64 + c];
        float s, co; __sincosf(-(TWO_PI / 32.f) * (float)(kyA * y), &s, &co);
        re += v.x * co - v.y * s; im += v.x * s + v.y * co;
    }
    Fy[t] = make_float2(re, im);
}

__global__ void k_dft_x(const float2* __restrict__ Fy, float2* __restrict__ Fx) {
    int t = blockIdx.x * blockDim.x + threadIdx.x;   // 16*16*8*64
    if (t >= 16 * 16 * 8 * 64) return;
    int c = t & 63, kz = (t >> 6) & 7, ky = (t >> 9) & 15, kx = t >> 13;
    int kxA = (kx < 8) ? kx : kx + 16;
    float re = 0.f, im = 0.f;
    for (int x = 0; x < 32; ++x) {
        float2 v = Fy[((size_t)(x * 16 + ky) * 8 + kz) * 64 + c];
        float s, co; __sincosf(-(TWO_PI / 32.f) * (float)(kxA * x), &s, &co);
        re += v.x * co - v.y * s; im += v.x * s + v.y * co;
    }
    Fx[t] = make_float2(re, im);
}

__global__ void k_einsum(const float2* __restrict__ Fx, const float* __restrict__ w,
                         float2* __restrict__ G) {
    int t = blockIdx.x * blockDim.x + threadIdx.x;   // 16*16*8*64
    if (t >= 16 * 16 * 8 * 64) return;
    int o = t & 63, kz = (t >> 6) & 7, ky = (t >> 9) & 15, kx = t >> 13;
    int corner = ((kx >= 8) ? 2 : 0) + ((ky >= 8) ? 1 : 0);
    int bx = kx & 7, by = ky & 7;
    const float2* fin = Fx + ((size_t)(kx * 16 + ky) * 8 + kz) * 64;
    const float* wb = w + ((((size_t)corner * 8 + bx) * 8 + by) * 8 + kz) * 64 * 64 * 2 + o * 2;
    float re = 0.f, im = 0.f;
    for (int i = 0; i < 64; ++i) {
        float a = fin[i].x, b = fin[i].y;
        float wr = wb[(size_t)i * 128], wi = wb[(size_t)i * 128 + 1];
        re += a * wr - b * wi; im += a * wi + b * wr;
    }
    G[t] = make_float2(re, im);
}

__global__ void k_idft_x(const float2* __restrict__ G, float2* __restrict__ Ix) {
    int t = blockIdx.x * blockDim.x + threadIdx.x;   // 32*16*8*64
    if (t >= 32 * 16 * 8 * 64) return;
    int c = t & 63, kz = (t >> 6) & 7, ky = (t >> 9) & 15, x = t >> 13;
    float re = 0.f, im = 0.f;
    for (int kx = 0; kx < 16; ++kx) {
        int kxA = (kx < 8) ? kx : kx + 16;
        float2 v = G[((size_t)(kx * 16 + ky) * 8 + kz) * 64 + c];
        float s, co; __sincosf((TWO_PI / 32.f) * (float)(kxA * x), &s, &co);
        re += v.x * co - v.y * s; im += v.x * s + v.y * co;
    }
    Ix[t] = make_float2(re, im);
}

__global__ void k_idft_y(const float2* __restrict__ Ix, float2* __restrict__ Iy) {
    int t = blockIdx.x * blockDim.x + threadIdx.x;   // 32*32*8*64
    if (t >= 32 * 32 * 8 * 64) return;
    int c = t & 63, kz = (t >> 6) & 7, y = (t >> 9) & 31, x = t >> 14;
    float re = 0.f, im = 0.f;
    for (int ky = 0; ky < 16; ++ky) {
        int kyA = (ky < 8) ? ky : ky + 16;
        float2 v = Ix[((size_t)(x * 16 + ky) * 8 + kz) * 64 + c];
        float s, co; __sincosf((TWO_PI / 32.f) * (float)(kyA * y), &s, &co);
        re += v.x * co - v.y * s; im += v.x * s + v.y * co;
    }
    Iy[t] = make_float2(re, im);
}

__global__ void k_idft_z(const float2* __restrict__ Iy, float* __restrict__ fno) {
    int t = blockIdx.x * blockDim.x + threadIdx.x;   // 32*32*32*64
    if (t >= 32 * 32 * 32 * 64) return;
    int c = t & 63, z = (t >> 6) & 31, y = (t >> 11) & 31, x = t >> 16;
    const float2* base = Iy + ((size_t)(x * 32 + y) * 8) * 64 + c;
    float acc = base[0].x;                           // kz = 0 term
    for (int kz = 1; kz < 8; ++kz) {
        float2 v = base[(size_t)kz * 64];
        float s, co; __sincosf((TWO_PI / 32.f) * (float)(kz * z), &s, &co);
        acc += 2.0f * (v.x * co - v.y * s);          // hermitian pair
    }
    fno[t] = acc * (1.0f / 32768.0f);                // irfftn normalization
}

// --------------------------- FNO block glue --------------------------------
__global__ void k_ew_a(const float* __restrict__ fno, const float* __restrict__ skip,
                       const float* __restrict__ gw, const float* __restrict__ gb,
                       _Float16* __restrict__ a16, float* __restrict__ gate, int n) {
    int t = blockIdx.x * blockDim.x + threadIdx.x;
    if (t >= n) return;
    int c = t & 63;
    float a = gelu_f(fno[t] + skip[t]);
    a16[t] = (_Float16)a;
    gate[t] = a * gw[c] + gb[c];
}

__global__ void k_ew_h(const float* __restrict__ m, const float* __restrict__ gate,
                       float* __restrict__ h32, _Float16* __restrict__ h16, int n) {
    int t = blockIdx.x * blockDim.x + threadIdx.x;
    if (t >= n) return;
    float v = gelu_f(m[t] + gate[t]);
    h32[t] = v; h16[t] = (_Float16)v;
}

// ===========================================================================
extern "C" void kernel_launch(void* const* d_in, const int* in_sizes, int n_in,
                              void* d_out, int out_size, void* d_ws, size_t ws_size,
                              hipStream_t stream) {
    const float* input_geom  = (const float*)d_in[0];
    const float* latent_q    = (const float*)d_in[1];
    const float* output_q    = (const float*)d_in[2];
    const float* xin         = (const float*)d_in[3];
    const float* in_lift_w   = (const float*)d_in[4];
    const float* in_lift_b   = (const float*)d_in[5];
    const float* in_k_w0 = (const float*)d_in[6];  const float* in_k_b0 = (const float*)d_in[7];
    const float* in_k_w1 = (const float*)d_in[8];  const float* in_k_b1 = (const float*)d_in[9];
    const float* in_k_w2 = (const float*)d_in[10]; const float* in_k_b2 = (const float*)d_in[11];
    const float* in_k_w3 = (const float*)d_in[12]; const float* in_k_b3 = (const float*)d_in[13];
    const float* lift_w0 = (const float*)d_in[14]; const float* lift_b0 = (const float*)d_in[15];
    const float* lift_w1 = (const float*)d_in[16]; const float* lift_b1 = (const float*)d_in[17];
    const float* spec_w  = (const float*)d_in[18];
    const float* skip_w  = (const float*)d_in[19];
    const float* gate_w  = (const float*)d_in[20]; const float* gate_b  = (const float*)d_in[21];
    const float* cmlp_w0 = (const float*)d_in[22]; const float* cmlp_b0 = (const float*)d_in[23];
    const float* cmlp_w1 = (const float*)d_in[24]; const float* cmlp_b1 = (const float*)d_in[25];
    const float* out_k_w0 = (const float*)d_in[26]; const float* out_k_b0 = (const float*)d_in[27];
    const float* out_k_w1 = (const float*)d_in[28]; const float* out_k_b1 = (const float*)d_in[29];
    const float* out_k_w2 = (const float*)d_in[30]; const float* out_k_b2 = (const float*)d_in[31];
    const float* proj_w0 = (const float*)d_in[32]; const float* proj_b0 = (const float*)d_in[33];
    const float* proj_w1 = (const float*)d_in[34]; const float* proj_b1 = (const float*)d_in[35];
    const int* in_idx  = (const int*)d_in[36];
    const int* out_idx = (const int*)d_in[37];

    const int N_IN = 100000, N_OUT = 30000, G3 = 32768;
    const int PCH = 4096;      // latent chunk (rows = 40960)
    const int QCH = 4000;      // query chunk  (rows = 40000 / 20000)

    size_t off = 0;
    auto alloc = [&](size_t bytes) -> void* {
        off = (off + 255) & ~(size_t)255;
        void* p = (char*)d_ws + off;
        off += bytes;
        return p;
    };

    _Float16* ye   = (_Float16*)alloc((size_t)N_IN * 96 * 2);
    _Float16* xe   = (_Float16*)alloc((size_t)G3 * 96 * 2);
    _Float16* oe   = (_Float16*)alloc((size_t)N_OUT * 96 * 2);
    float*    fbuf = (float*)alloc((size_t)N_IN * 64 * 4);
    float*    h32  = (float*)alloc((size_t)G3 * 64 * 4);
    _Float16* h16  = (_Float16*)alloc((size_t)G3 * 64 * 2);

    // packed f16 weights (WMMA B-fragment order)
    _Float16* ikw0 = (_Float16*)alloc(192 * 96 * 2);
    _Float16* ikw1 = (_Float16*)alloc(96 * 96 * 2);
    _Float16* ikw2 = (_Float16*)alloc(96 * 96 * 2);
    _Float16* ikw3 = (_Float16*)alloc(96 * 64 * 2);
    float* ikb0p = (float*)alloc(96 * 4);
    float* ikb1p = (float*)alloc(96 * 4);
    float* ikb2p = (float*)alloc(96 * 4);
    _Float16* lw0 = (_Float16*)alloc(64 * 128 * 2);
    _Float16* lw1 = (_Float16*)alloc(128 * 64 * 2);
    _Float16* skwh = (_Float16*)alloc(4 * 64 * 64 * 2);
    _Float16* cm0h = (_Float16*)alloc(4 * 64 * 128 * 2);
    _Float16* cm1h = (_Float16*)alloc(4 * 128 * 64 * 2);
    _Float16* okw0 = (_Float16*)alloc(192 * 512 * 2);
    _Float16* okw1 = (_Float16*)alloc(512 * 256 * 2);
    _Float16* okw2 = (_Float16*)alloc(256 * 64 * 2);
    _Float16* pw0h = (_Float16*)alloc(64 * 256 * 2);
    _Float16* pw1h = (_Float16*)alloc(256 * 32 * 2);
    float* pb1p = (float*)alloc(32 * 4);

    // spectral scratch
    float2* Fz = (float2*)alloc((size_t)32 * 32 * 8 * 64 * 8);
    float2* Fy = (float2*)alloc((size_t)32 * 16 * 8 * 64 * 8);
    float2* Fx = (float2*)alloc((size_t)16 * 16 * 8 * 64 * 8);
    float2* Gm = (float2*)alloc((size_t)16 * 16 * 8 * 64 * 8);
    float2* Ix = (float2*)alloc((size_t)32 * 16 * 8 * 64 * 8);
    float2* Iy = (float2*)alloc((size_t)32 * 32 * 8 * 64 * 8);
    float*  fno   = (float*)alloc((size_t)G3 * 64 * 4);
    float*  skipv = (float*)alloc((size_t)G3 * 64 * 4);
    _Float16* a16 = (_Float16*)alloc((size_t)G3 * 64 * 2);
    float*  gate  = (float*)alloc((size_t)G3 * 64 * 4);
    _Float16* t16 = (_Float16*)alloc((size_t)G3 * 128 * 2);
    float*  m32   = (float*)alloc((size_t)G3 * 64 * 4);

    // input-GNO chunk arena (40960 rows)
    _Float16* EFi = (_Float16*)alloc((size_t)PCH * 10 * 192 * 2);
    _Float16* Ha  = (_Float16*)alloc((size_t)PCH * 10 * 96 * 2);
    _Float16* Hb  = (_Float16*)alloc((size_t)PCH * 10 * 96 * 2);
    float*    K3i = (float*)alloc((size_t)PCH * 10 * 64 * 4);

    // output-GNO chunk arena (40000 rows)
    _Float16* EFo  = (_Float16*)alloc((size_t)QCH * 10 * 192 * 2);
    _Float16* H512 = (_Float16*)alloc((size_t)QCH * 10 * 512 * 2);
    _Float16* H256 = (_Float16*)alloc((size_t)QCH * 10 * 256 * 2);
    float*    K3o  = (float*)alloc((size_t)QCH * 10 * 64 * 4);
    _Float16* oev16 = (_Float16*)alloc((size_t)N_OUT * 64 * 2);
    _Float16* tproj = (_Float16*)alloc((size_t)N_OUT * 256 * 2);

    auto gemm = [&](const _Float16* A, const _Float16* Wp, const float* bias,
                    _Float16* Ch, float* Cf, int M, int N, int Kd, int n_true, int dogelu) {
        dim3 g((M + 127) / 128, N / 32);
        k_gemm<<<g, 128, 0, stream>>>(A, Wp, bias, Ch, Cf, M, N, Kd, n_true, dogelu);
    };
    auto B = [](int n) { return (n + 255) / 256; };

    // ---- weight packing (fragment order + zero padding) ----
    k_packw<<<B(192 * 96), 256, 0, stream>>>(in_k_w0, ikw0, 192, 80, 192, 96);
    k_packw<<<B(96 * 96), 256, 0, stream>>>(in_k_w1, ikw1, 80, 80, 96, 96);
    k_packw<<<B(96 * 96), 256, 0, stream>>>(in_k_w2, ikw2, 80, 80, 96, 96);
    k_packw<<<B(96 * 64), 256, 0, stream>>>(in_k_w3, ikw3, 80, 64, 96, 64);
    k_padb<<<1, 96, 0, stream>>>(in_k_b0, ikb0p, 80, 96);
    k_padb<<<1, 96, 0, stream>>>(in_k_b1, ikb1p, 80, 96);
    k_padb<<<1, 96, 0, stream>>>(in_k_b2, ikb2p, 80, 96);
    k_packw<<<B(64 * 128), 256, 0, stream>>>(lift_w0, lw0, 64, 128, 64, 128);
    k_packw<<<B(128 * 64), 256, 0, stream>>>(lift_w1, lw1, 128, 64, 128, 64);
    // layer-stacked weights: 64/128 rows = 2/4 kchunks -> packed layers stay at l*K*N
    k_packw<<<B(256 * 64), 256, 0, stream>>>(skip_w, skwh, 256, 64, 256, 64);
    k_packw<<<B(256 * 128), 256, 0, stream>>>(cmlp_w0, cm0h, 256, 128, 256, 128);
    k_packw<<<B(512 * 64), 256, 0, stream>>>(cmlp_w1, cm1h, 512, 64, 512, 64);
    k_packw<<<B(192 * 512), 256, 0, stream>>>(out_k_w0, okw0, 192, 512, 192, 512);
    k_packw<<<B(512 * 256), 256, 0, stream>>>(out_k_w1, okw1, 512, 256, 512, 256);
    k_packw<<<B(256 * 64), 256, 0, stream>>>(out_k_w2, okw2, 256, 64, 256, 64);
    k_packw<<<B(64 * 256), 256, 0, stream>>>(proj_w0, pw0h, 64, 256, 64, 256);
    k_packw<<<B(256 * 32), 256, 0, stream>>>(proj_w1, pw1h, 256, 1, 256, 32);
    k_padb<<<1, 32, 0, stream>>>(proj_b1, pb1p, 1, 32);

    // ---- embeddings + input lift ----
    k_embed<<<B(N_IN * 48), 256, 0, stream>>>(input_geom, ye, N_IN);
    k_embed<<<B(G3 * 48), 256, 0, stream>>>(latent_q, xe, G3);
    k_embed<<<B(N_OUT * 48), 256, 0, stream>>>(output_q, oe, N_OUT);
    k_lift<<<B(N_IN * 64), 256, 0, stream>>>(xin, in_lift_w, in_lift_b, fbuf, N_IN);

    // ---- input GNO (chunked over latent points): 192->96->96->96->64, mean ----
    for (int p0 = 0; p0 < G3; p0 += PCH) {
        int rows = PCH * 10;
        k_gather_in<<<B(rows * 192), 256, 0, stream>>>(ye, xe, in_idx, EFi, p0, PCH);
        gemm(EFi, ikw0, ikb0p, Ha, nullptr, rows, 96, 192, 96, 1);
        gemm(Ha, ikw1, ikb1p, Hb, nullptr, rows, 96, 96, 96, 1);
        gemm(Hb, ikw2, ikb2p, Ha, nullptr, rows, 96, 96, 96, 1);
        gemm(Ha, ikw3, in_k_b3, nullptr, K3i, rows, 64, 96, 64, 0);
        k_reduce_in<<<B(PCH * 64), 256, 0, stream>>>(K3i, fbuf, in_idx, h32, h16, p0, PCH);
    }

    // ---- lifting ChannelMLP: 64 -> 128 (gelu) -> 64 ----
    gemm(h16, lw0, lift_b0, t16, nullptr, G3, 128, 64, 128, 1);
    gemm(t16, lw1, lift_b1, h16, h32, G3, 64, 128, 64, 0);

    // ---- FNO blocks ----
    for (int l = 0; l < 4; ++l) {
        k_dft_z<<<B(32 * 32 * 8 * 64), 256, 0, stream>>>(h32, Fz);
        k_dft_y<<<B(32 * 16 * 8 * 64), 256, 0, stream>>>(Fz, Fy);
        k_dft_x<<<B(16 * 16 * 8 * 64), 256, 0, stream>>>(Fy, Fx);
        k_einsum<<<B(16 * 16 * 8 * 64), 256, 0, stream>>>(
            Fx, spec_w + (size_t)l * 4 * 8 * 8 * 8 * 64 * 64 * 2, Gm);
        k_idft_x<<<B(32 * 16 * 8 * 64), 256, 0, stream>>>(Gm, Ix);
        k_idft_y<<<B(32 * 32 * 8 * 64), 256, 0, stream>>>(Ix, Iy);
        k_idft_z<<<B(32 * 32 * 32 * 64), 256, 0, stream>>>(Iy, fno);
        gemm(h16, skwh + (size_t)l * 64 * 64, nullptr, nullptr, skipv, G3, 64, 64, 64, 0);
        k_ew_a<<<B(G3 * 64), 256, 0, stream>>>(fno, skipv, gate_w + l * 64, gate_b + l * 64,
                                               a16, gate, G3 * 64);
        gemm(a16, cm0h + (size_t)l * 64 * 128, cmlp_b0 + l * 128, t16, nullptr, G3, 128, 64, 128, 1);
        gemm(t16, cm1h + (size_t)l * 128 * 64, cmlp_b1 + l * 64, nullptr, m32, G3, 64, 128, 64, 0);
        k_ew_h<<<B(G3 * 64), 256, 0, stream>>>(m32, gate, h32, h16, G3 * 64);
    }

    // ---- output GNO (chunked over queries): 192->512->256->64, sum ----
    for (int q0 = 0; q0 < N_OUT; q0 += QCH) {
        int qn = (N_OUT - q0 < QCH) ? (N_OUT - q0) : QCH;
        int rows = qn * 10;
        k_gather_out<<<B(rows * 192), 256, 0, stream>>>(xe, oe, out_idx, EFo, q0, qn);
        gemm(EFo, okw0, out_k_b0, H512, nullptr, rows, 512, 192, 512, 1);
        gemm(H512, okw1, out_k_b1, H256, nullptr, rows, 256, 512, 256, 1);
        gemm(H256, okw2, out_k_b2, nullptr, K3o, rows, 64, 256, 64, 0);
        k_reduce_out<<<B(qn * 64), 256, 0, stream>>>(K3o, h32, out_idx, oev16, q0, qn);
    }

    // ---- projection: 64 -> 256 (gelu) -> 1 ----
    gemm(oev16, pw0h, proj_b0, tproj, nullptr, N_OUT, 256, 64, 256, 1);
    gemm(tproj, pw1h, pb1p, nullptr, (float*)d_out, N_OUT, 32, 256, 1, 0);
}
